// LinearAttention_57767310131390
// MI455X (gfx1250) — matrix operand order repaired and checked by reading, not compiled, not run on hardware
//
#include <hip/hip_runtime.h>

typedef __bf16 bf16_t;
typedef __attribute__((ext_vector_type(16))) __bf16 bf16x16;
typedef __attribute__((ext_vector_type(8)))  float   f32x8;
typedef __attribute__((ext_vector_type(4)))  int     i32x4;

#define R_ROWS 25088   // B*N = 128*196
#define C_DIM  768
#define NH     12
#define EH     64
#define S_LEN  3136    // 16*196
#define NBH    96      // (128/16)*12
#define EPS_F  0.0001f

// ---------------- f32 -> bf16 conversion ----------------
__global__ void cvt_f32_to_bf16(const float* __restrict__ in,
                                bf16_t* __restrict__ out, int n) {
    int i = blockIdx.x * blockDim.x + threadIdx.x;
    if (i < n) out[i] = (bf16_t)in[i];
}

// ---------------- bf16 WMMA GEMM:  C[M,N] = A[M,K] * W[N,K]^T ----------------
// mode 0: C = f32, += bias[col]
// mode 1: C = bf16, relu applied to cols < 1536 (q,k of qkv)
__global__ __launch_bounds__(256) void gemm_bf16_wmma(
    const bf16_t* __restrict__ A,
    const bf16_t* __restrict__ W,
    const float*  __restrict__ bias,
    void* __restrict__ Cout,
    int M, int N, int K, int mode)
{
    const int ntiles = N >> 7;                 // N / 128
    const int bm = blockIdx.x / ntiles;
    const int bn = blockIdx.x % ntiles;
    const int wave  = threadIdx.x >> 5;
    const int lane  = threadIdx.x & 31;
    const int wm    = wave & 3;                // 4 waves along M
    const int wn    = wave >> 2;               // 2 waves along N
    const int lhalf = lane >> 4;               // lane group 0/1
    const int lmod  = lane & 15;
    const int row0  = bm * 128 + wm * 32;      // 2 x 16-row tiles
    const int col0  = bn * 128 + wn * 64;      // 4 x 16-col tiles

    const bf16_t* arow0 = A + (size_t)(row0 + lmod) * K;
    const bf16_t* arow1 = arow0 + (size_t)16 * K;
    const bf16_t* brow[4];
#pragma unroll
    for (int ni = 0; ni < 4; ++ni)
        brow[ni] = W + (size_t)(col0 + ni * 16 + lmod) * K;

    f32x8 acc[2][4];
#pragma unroll
    for (int mi = 0; mi < 2; ++mi)
#pragma unroll
        for (int ni = 0; ni < 4; ++ni)
#pragma unroll
            for (int r = 0; r < 8; ++r) acc[mi][ni][r] = 0.0f;

    union Frag { bf16x16 v; i32x4 q[2]; };

    for (int kb = 0; kb < K; kb += 32) {
        Frag af[2], bfr[4];
        // A frag: halves[0..7]=A[m][kb+lh*8..], halves[8..15]=A[m][kb+16+lh*8..]
        {
            const bf16_t* p = arow0 + kb + lhalf * 8;
            af[0].q[0] = *(const i32x4*)(p);
            af[0].q[1] = *(const i32x4*)(p + 16);
            p = arow1 + kb + lhalf * 8;
            af[1].q[0] = *(const i32x4*)(p);
            af[1].q[1] = *(const i32x4*)(p + 16);
        }
        // B frag: halves[0..7]=W[n][kb+lh*8..], halves[8..15]=W[n][kb+16+lh*8..]
#pragma unroll
        for (int ni = 0; ni < 4; ++ni) {
            const bf16_t* p = brow[ni] + kb + lhalf * 8;
            bfr[ni].q[0] = *(const i32x4*)(p);
            bfr[ni].q[1] = *(const i32x4*)(p + 16);
        }
#pragma unroll
        for (int mi = 0; mi < 2; ++mi)
#pragma unroll
            for (int ni = 0; ni < 4; ++ni)
                acc[mi][ni] = __builtin_amdgcn_wmma_f32_16x16x32_bf16(
                    false, af[mi].v, false, bfr[ni].v,
                    (short)0, acc[mi][ni], false, false);
    }

    // C/D layout: vgpr r, lanes 0-15 -> M=r, lanes 16-31 -> M=r+8; N = lane%16
    if (mode == 0) {
        float* Cf = (float*)Cout;
#pragma unroll
        for (int mi = 0; mi < 2; ++mi)
#pragma unroll
            for (int ni = 0; ni < 4; ++ni) {
                const int col = col0 + ni * 16 + lmod;
                const float bv = bias ? bias[col] : 0.0f;
#pragma unroll
                for (int r = 0; r < 8; ++r) {
                    const int row = row0 + mi * 16 + lhalf * 8 + r;
                    Cf[(size_t)row * N + col] = acc[mi][ni][r] + bv;
                }
            }
    } else {
        bf16_t* Cb = (bf16_t*)Cout;
#pragma unroll
        for (int mi = 0; mi < 2; ++mi)
#pragma unroll
            for (int ni = 0; ni < 4; ++ni) {
                const int col = col0 + ni * 16 + lmod;
                const bool do_relu = (col < 1536);   // q and k columns
#pragma unroll
                for (int r = 0; r < 8; ++r) {
                    const int row = row0 + mi * 16 + lhalf * 8 + r;
                    float v = acc[mi][ni][r];
                    if (do_relu && v < 0.0f) v = 0.0f;
                    Cb[(size_t)row * N + col] = (bf16_t)v;
                }
            }
    }
}

// ---------------- kv[b,h,e,d] = sum_s k[s,e]*v[s,d] ; ksum[b,h,e] ----------------
__global__ __launch_bounds__(256) void kv_ksum_kernel(
    const bf16_t* __restrict__ qkv,   // R x 2304 (q|k|v)
    float* __restrict__ kvout,        // NBH x 64 x 64
    float* __restrict__ ksum)         // NBH x 64
{
    const int bh = blockIdx.x;
    const int b  = bh / NH, h = bh % NH;
    const int t  = threadIdx.x;
    const int e    = t >> 2;          // 0..63
    const int dgrp = t & 3;           // 16 d-values each
    __shared__ bf16_t kt[64 * 64];
    __shared__ bf16_t vt[64 * 64];

    float acc[16];
#pragma unroll
    for (int j = 0; j < 16; ++j) acc[j] = 0.0f;
    float ks = 0.0f;

    const int sl = t >> 2, part = t & 3;   // each thread stages 16 elems of row sl
    for (int sc = 0; sc < S_LEN / 64; ++sc) {
        const size_t rowbase = (size_t)(b * S_LEN + sc * 64 + sl) * 2304;
        const bf16_t* ksrc = qkv + rowbase + 768  + h * 64 + part * 16;
        const bf16_t* vsrc = qkv + rowbase + 1536 + h * 64 + part * 16;
        i32x4* kdst = (i32x4*)&kt[sl * 64 + part * 16];
        i32x4* vdst = (i32x4*)&vt[sl * 64 + part * 16];
        kdst[0] = ((const i32x4*)ksrc)[0];
        kdst[1] = ((const i32x4*)ksrc)[1];
        vdst[0] = ((const i32x4*)vsrc)[0];
        vdst[1] = ((const i32x4*)vsrc)[1];
        __syncthreads();
        for (int s = 0; s < 64; ++s) {
            const float kval = (float)kt[s * 64 + e];
            if (dgrp == 0) ks += kval;
            const bf16_t* vrow = &vt[s * 64 + dgrp * 16];
#pragma unroll
            for (int j = 0; j < 16; ++j) acc[j] += kval * (float)vrow[j];
        }
        __syncthreads();
    }
    float* kvo = kvout + (size_t)bh * 4096 + e * 64 + dgrp * 16;
#pragma unroll
    for (int j = 0; j < 16; ++j) kvo[j] = acc[j];
    if (dgrp == 0) ksum[bh * 64 + e] = ks;
}

// ---------------- out[s,h,d] = (q . kv) / (q . ksum + eps), bf16 ----------------
__global__ __launch_bounds__(256) void attn_out_kernel(
    const bf16_t* __restrict__ qkv,
    const float*  __restrict__ kvmat,
    const float*  __restrict__ ksum,
    bf16_t* __restrict__ attn)        // R x 768
{
    const int chunk = blockIdx.x;       // 64-row chunk; 392 total
    const int h     = blockIdx.y;
    const int row0  = chunk * 64;
    const int b     = row0 / S_LEN;     // chunks never straddle b (3136 % 64 == 0)
    const int bh    = b * NH + h;
    const int t = threadIdx.x;
    __shared__ float  kvs[64 * 64];
    __shared__ float  kss[64];
    __shared__ bf16_t qt[64 * 64];

    for (int i = t; i < 4096; i += 256) kvs[i] = kvmat[(size_t)bh * 4096 + i];
    if (t < 64) kss[t] = ksum[bh * 64 + t];
    {
        const int sl = t >> 2, part = t & 3;
        const bf16_t* qsrc = qkv + (size_t)(row0 + sl) * 2304 + h * 64 + part * 16;
        i32x4* qdst = (i32x4*)&qt[sl * 64 + part * 16];
        qdst[0] = ((const i32x4*)qsrc)[0];
        qdst[1] = ((const i32x4*)qsrc)[1];
    }
    __syncthreads();

    const int r = t >> 2, dgrp = t & 3;
    float zden = EPS_F;
    for (int e = 0; e < 64; ++e) zden += (float)qt[r * 64 + e] * kss[e];
    const float z = 1.0f / zden;

    float o[16];
#pragma unroll
    for (int j = 0; j < 16; ++j) o[j] = 0.0f;
    for (int e = 0; e < 64; ++e) {
        const float qv = (float)qt[r * 64 + e];
        const float* krow = &kvs[e * 64 + dgrp * 16];
#pragma unroll
        for (int j = 0; j < 16; ++j) o[j] += qv * krow[j];
    }
    bf16_t* dst = attn + (size_t)(row0 + r) * C_DIM + h * 64 + dgrp * 16;
#pragma unroll
    for (int j = 0; j < 16; ++j) dst[j] = (bf16_t)(o[j] * z);
}

// ---------------- launcher ----------------
extern "C" void kernel_launch(void* const* d_in, const int* in_sizes, int n_in,
                              void* d_out, int out_size, void* d_ws, size_t ws_size,
                              hipStream_t stream) {
    const float* x     = (const float*)d_in[0];
    const float* Wqkv  = (const float*)d_in[1];
    const float* Wproj = (const float*)d_in[2];
    const float* bproj = (const float*)d_in[3];

    char* ws = (char*)d_ws;
    size_t off = 0;
    auto alloc = [&](size_t bytes) -> void* {
        void* p = ws + off;
        off = (off + bytes + 255) & ~(size_t)255;
        return p;
    };
    bf16_t* xb     = (bf16_t*)alloc((size_t)R_ROWS * C_DIM * 2);
    bf16_t* wqkvb  = (bf16_t*)alloc((size_t)3 * C_DIM * C_DIM * 2);
    bf16_t* wprojb = (bf16_t*)alloc((size_t)C_DIM * C_DIM * 2);
    bf16_t* qkvb   = (bf16_t*)alloc((size_t)R_ROWS * 3 * C_DIM * 2);
    bf16_t* attnb  = (bf16_t*)alloc((size_t)R_ROWS * C_DIM * 2);
    float*  kvw    = (float*)alloc((size_t)NBH * 64 * 64 * 4);
    float*  ksw    = (float*)alloc((size_t)NBH * 64 * 4);

    // 1) precision conversion
    {
        int n = R_ROWS * C_DIM;
        cvt_f32_to_bf16<<<(n + 255) / 256, 256, 0, stream>>>(x, xb, n);
        n = 3 * C_DIM * C_DIM;
        cvt_f32_to_bf16<<<(n + 255) / 256, 256, 0, stream>>>(Wqkv, wqkvb, n);
        n = C_DIM * C_DIM;
        cvt_f32_to_bf16<<<(n + 255) / 256, 256, 0, stream>>>(Wproj, wprojb, n);
    }
    // 2) QKV GEMM (25088x768 @ 768x2304^T), relu(q,k) epilogue, bf16 out
    gemm_bf16_wmma<<<(R_ROWS / 128) * (2304 / 128), 256, 0, stream>>>(
        xb, wqkvb, nullptr, qkvb, R_ROWS, 2304, C_DIM, /*mode=*/1);
    // 3) kv = k^T v and k_sum per (b,h)
    kv_ksum_kernel<<<NBH, 256, 0, stream>>>(qkvb, kvw, ksw);
    // 4) out = (q @ kv) * 1/(q.k_sum + eps), bf16
    attn_out_kernel<<<dim3(R_ROWS / 64, NH), 256, 0, stream>>>(qkvb, kvw, ksw, attnb);
    // 5) proj GEMM (25088x768 @ 768x768^T) + bias, f32 out
    gemm_bf16_wmma<<<(R_ROWS / 128) * (C_DIM / 128), 256, 0, stream>>>(
        attnb, wprojb, bproj, d_out, R_ROWS, C_DIM, C_DIM, /*mode=*/0);
}
